// NeighborhoodConvAttention_26963804684922
// MI455X (gfx1250) — compile-verified, matrix operand order checked by reading, not run on hardware
//
#include <hip/hip_runtime.h>
#include <hip/hip_bf16.h>

// ---------------------------------------------------------------------------
// NeighborhoodConvAttention for MI455X (gfx1250, wave32, WMMA)
// B=2, C=128, H=W=64, HEADS=4, DHEAD=32, K=7, GROUPS=32
// ---------------------------------------------------------------------------

#define Bsz 2
#define Cch 128
#define Hh 64
#define Ww 64
#define HEADS 4
#define DHEAD 32
#define DA 128
#define KW 7
#define NSPAT (Bsz * Hh * Ww)   // 8192 spatial positions
#define EPSV 1e-5f

typedef __attribute__((ext_vector_type(16))) _Float16 v16h;
typedef __attribute__((ext_vector_type(8)))  _Float16 v8h_t;
typedef __attribute__((ext_vector_type(4)))  _Float16 v4h;
typedef __attribute__((ext_vector_type(8)))  float    v8f;

// ---------------------------------------------------------------------------
// WMMA operand load: 16-bit 16x32 operand, per documented lane layout:
//   lane L: row = L & 15, k-chunks [kb, kb+8) and [kb+16, kb+24), kb=(L>>4)*8
// A is M x K row-major; B is supplied as N x K row-major (weight layout),
// which mirrors the A per-lane pattern.
// ---------------------------------------------------------------------------
__device__ __forceinline__ v16h load_nk_tile(const _Float16* base, int ld, int lane) {
    int row = lane & 15;
    int kb  = (lane >> 4) << 3;               // 0 or 8
    const _Float16* p = base + (size_t)row * ld + kb;
    v8h_t lo = *(const v8h_t*)(p);            // 16-byte load
    v8h_t hi = *(const v8h_t*)(p + 16);       // 16-byte load
    v16h r;
#pragma unroll
    for (int i = 0; i < 8; ++i) { r[i] = lo[i]; r[i + 8] = hi[i]; }
    return r;
}

__device__ __forceinline__ v8f wmma_step(v16h a, v16h b, v8f c) {
    return __builtin_amdgcn_wmma_f32_16x16x32_f16(
        /*neg_a=*/false, a, /*neg_b=*/false, b,
        /*c_mod=*/(short)0, c, /*reuse_a=*/false, /*reuse_b=*/false);
}

// ---------------------------------------------------------------------------
// Kernel 0: convert the four 128x128 fp32 weight matrices to f16 (once)
// ---------------------------------------------------------------------------
__global__ void convert_weights_kernel(const float* __restrict__ wq, const float* __restrict__ wk,
                                       const float* __restrict__ wv, const float* __restrict__ wo,
                                       _Float16* __restrict__ wqh, _Float16* __restrict__ wkh,
                                       _Float16* __restrict__ wvh, _Float16* __restrict__ woh) {
    int i = blockIdx.x * 256 + threadIdx.x;   // 16384 threads total
    wqh[i] = (_Float16)wq[i];
    wkh[i] = (_Float16)wk[i];
    wvh[i] = (_Float16)wv[i];
    woh[i] = (_Float16)wo[i];
}

// ---------------------------------------------------------------------------
// Kernel 1: GroupNorm.  One block per (batch, group); group = 4 consecutive
// channels, 16384 elements.  Writes normalized activation as f16 in
// [spatial, C] layout; one thread normalizes all 4 channels of a spatial
// point -> single 8-byte packed store.
// ---------------------------------------------------------------------------
__global__ void groupnorm_kernel(const float* __restrict__ x, const float* __restrict__ gw,
                                 const float* __restrict__ gb, _Float16* __restrict__ xn) {
    int bg = blockIdx.x;               // 0..63
    int b  = bg >> 5;                  // batch
    int g  = bg & 31;                  // group
    int tid = threadIdx.x;             // 256 threads
    const float* xp = x + ((size_t)b * Cch + g * 4) * (Hh * Ww);   // 4 channels * 4096

    float s = 0.f, ss = 0.f;
    for (int i = tid; i < 16384; i += 256) {
        float v = xp[i];
        s += v; ss += v * v;
    }
    __shared__ float sh0[256], sh1[256];
    sh0[tid] = s; sh1[tid] = ss;
    __syncthreads();
    for (int st = 128; st > 0; st >>= 1) {
        if (tid < st) { sh0[tid] += sh0[tid + st]; sh1[tid] += sh1[tid + st]; }
        __syncthreads();
    }
    float mean = sh0[0] * (1.f / 16384.f);
    float var  = sh1[0] * (1.f / 16384.f) - mean * mean;
    float rstd = rsqrtf(var + EPSV);

    float w0 = gw[g * 4 + 0] * rstd, w1 = gw[g * 4 + 1] * rstd;
    float w2 = gw[g * 4 + 2] * rstd, w3 = gw[g * 4 + 3] * rstd;
    float b0 = gb[g * 4 + 0], b1 = gb[g * 4 + 1];
    float b2 = gb[g * 4 + 2], b3 = gb[g * 4 + 3];

    for (int hw = tid; hw < 4096; hw += 256) {
        float v0 = (xp[hw            ] - mean) * w0 + b0;
        float v1 = (xp[hw +  1 * 4096] - mean) * w1 + b1;
        float v2 = (xp[hw +  2 * 4096] - mean) * w2 + b2;
        float v3 = (xp[hw +  3 * 4096] - mean) * w3 + b3;
        v4h pk = { (_Float16)v0, (_Float16)v1, (_Float16)v2, (_Float16)v3 };
        *(v4h*)(xn + ((size_t)b * 4096 + hw) * 128 + g * 4) = pk;   // 8B store
    }
}

// ---------------------------------------------------------------------------
// Kernel 2: Q/K/V projections via WMMA.  8 waves/block; each wave computes a
// full 16(M) x 128(N) row-block of one projection.  The A row-block (16x128,
// 4 chunks = 32 VGPRs) is loaded ONCE; then each 16x16 N-tile is computed
// (4 WMMAs) and stored immediately, so only one v8f accumulator is live --
// this leaves registers free for the scheduler to pipeline B-tile loads
// across N-tiles instead of serializing on s_wait_loadcnt 0.
// Jobs: 512 M-tiles x 3 projections = 1536 waves -> 192 blocks.
// ---------------------------------------------------------------------------
__global__ __launch_bounds__(256, 1)
void qkv_proj_kernel(const _Float16* __restrict__ xn,
                     const _Float16* __restrict__ wqh, const _Float16* __restrict__ wkh,
                     const _Float16* __restrict__ wvh,
                     const float* __restrict__ bq, const float* __restrict__ bk,
                     const float* __restrict__ bv,
                     float* __restrict__ qb, float* __restrict__ kb, float* __restrict__ vb) {
    int wid  = threadIdx.x >> 5;
    int lane = threadIdx.x & 31;
    int t    = blockIdx.x * 8 + wid;        // 0..1535
    int proj = t >> 9;                      // 512 M-tiles per projection
    int mt   = t & 511;

    const _Float16* W   = (proj == 0) ? wqh : (proj == 1) ? wkh : wvh;
    const float*    bia = (proj == 0) ? bq  : (proj == 1) ? bk  : bv;
    float*          dst = (proj == 0) ? qb  : (proj == 1) ? kb  : vb;
    float scale = (proj == 0) ? 0.17677669529663687f : 1.0f;   // 1/sqrt(32) for Q

    const _Float16* A = xn + (size_t)mt * 16 * 128;
    v16h a[4];
#pragma unroll
    for (int kk = 0; kk < 4; ++kk) a[kk] = load_nk_tile(A + kk * 32, 128, lane);

    int n = lane & 15;
    int mbase = (lane >> 4) << 3;

#pragma unroll
    for (int nt = 0; nt < 8; ++nt) {
        const _Float16* Bt = W + (size_t)nt * 16 * 128;
        v16h b0 = load_nk_tile(Bt +  0, 128, lane);
        v16h b1 = load_nk_tile(Bt + 32, 128, lane);
        v16h b2 = load_nk_tile(Bt + 64, 128, lane);
        v16h b3 = load_nk_tile(Bt + 96, 128, lane);
        v8f c = {};
        c = wmma_step(a[0], b0, c);
        c = wmma_step(a[1], b1, c);
        c = wmma_step(a[2], b2, c);
        c = wmma_step(a[3], b3, c);

        int o = nt * 16 + n;                // output channel (head*32 + d)
        float bs = bia[o];
#pragma unroll
        for (int r = 0; r < 8; ++r) {
            int s = mt * 16 + mbase + r;    // spatial index
            dst[(size_t)s * 128 + o] = (c[r] + bs) * scale;
        }
    }
}

// ---------------------------------------------------------------------------
// Kernel 3: neighborhood attention, flash-style online softmax.
// One thread per query (32768 queries).  Window clamping keeps all 49 keys
// inside the image -> no masking.  d=32 via float4 loads.
// Writes f16 xo[s*128 + head*32 + d] (A operand of out-proj).
// ---------------------------------------------------------------------------
__global__ void natt_kernel(const float* __restrict__ qb, const float* __restrict__ kb,
                            const float* __restrict__ vb, const float* __restrict__ rpb,
                            _Float16* __restrict__ xo) {
    int t = blockIdx.x * 256 + threadIdx.x;     // 0..32767
    int qc   = t & 63;
    int p    = (t >> 6) & 63;
    int head = (t >> 12) & 3;
    int b    = t >> 14;

    size_t s = ((size_t)b * Hh + p) * Ww + qc;
    const float4* qp = (const float4*)(qb + s * 128 + head * 32);
    float4 qv[8];
#pragma unroll
    for (int u = 0; u < 8; ++u) qv[u] = qp[u];

    // window starts (clamped) and bias starts
    int hs  = min(max(p  - 3, 0), Hh - KW);
    int ws0 = min(max(qc - 3, 0), Ww - KW);
    int bh0 = 3 + max(3 - p, 0)  + min(Hh - p - 4, 0);
    int bw0 = 3 + max(3 - qc, 0) + min(Ww - qc - 4, 0);

    float m = -1e30f, l = 0.f;
    float4 acc[8];
#pragma unroll
    for (int u = 0; u < 8; ++u) acc[u] = make_float4(0.f, 0.f, 0.f, 0.f);

    for (int i = 0; i < KW; ++i) {
        int row = hs + i;
        const float* brow = rpb + head * (2 * KW - 1) * (2 * KW - 1) + (bh0 + i) * (2 * KW - 1) + bw0;
        for (int j = 0; j < KW; ++j) {
            int col = ws0 + j;
            size_t sn = ((size_t)b * Hh + row) * Ww + col;
            const float4* kp = (const float4*)(kb + sn * 128 + head * 32);
            float sc = brow[j];
#pragma unroll
            for (int u = 0; u < 8; ++u) {
                float4 kk = kp[u];
                sc += qv[u].x * kk.x + qv[u].y * kk.y + qv[u].z * kk.z + qv[u].w * kk.w;
            }
            float mn   = fmaxf(m, sc);
            float corr = __expf(m - mn);
            float pe   = __expf(sc - mn);
            l = l * corr + pe;
            const float4* vp = (const float4*)(vb + sn * 128 + head * 32);
#pragma unroll
            for (int u = 0; u < 8; ++u) {
                float4 vv = vp[u];
                acc[u].x = acc[u].x * corr + pe * vv.x;
                acc[u].y = acc[u].y * corr + pe * vv.y;
                acc[u].z = acc[u].z * corr + pe * vv.z;
                acc[u].w = acc[u].w * corr + pe * vv.w;
            }
            m = mn;
        }
    }

    float inv = 1.0f / l;
    _Float16* op = xo + s * 128 + head * 32;
#pragma unroll
    for (int u = 0; u < 8; ++u) {
        v4h pk = { (_Float16)(acc[u].x * inv), (_Float16)(acc[u].y * inv),
                   (_Float16)(acc[u].z * inv), (_Float16)(acc[u].w * inv) };
        *(v4h*)(op + 4 * u) = pk;
    }
}

// ---------------------------------------------------------------------------
// Kernel 4: output projection via WMMA.  Same per-wave structure as the QKV
// kernel (A loaded once, tile-at-a-time completion).  512 waves -> 64 blocks.
// Scatters results into (B, C, H, W) fp32 output.
// ---------------------------------------------------------------------------
__global__ __launch_bounds__(256, 1)
void out_proj_kernel(const _Float16* __restrict__ xo, const _Float16* __restrict__ woh,
                     const float* __restrict__ bo, float* __restrict__ out) {
    int wid  = threadIdx.x >> 5;
    int lane = threadIdx.x & 31;
    int mt   = blockIdx.x * 8 + wid;        // 0..511

    const _Float16* A = xo + (size_t)mt * 16 * 128;
    v16h a[4];
#pragma unroll
    for (int kk = 0; kk < 4; ++kk) a[kk] = load_nk_tile(A + kk * 32, 128, lane);

    int n = lane & 15;
    int mbase = (lane >> 4) << 3;

#pragma unroll
    for (int nt = 0; nt < 8; ++nt) {
        const _Float16* Bt = woh + (size_t)nt * 16 * 128;
        v16h b0 = load_nk_tile(Bt +  0, 128, lane);
        v16h b1 = load_nk_tile(Bt + 32, 128, lane);
        v16h b2 = load_nk_tile(Bt + 64, 128, lane);
        v16h b3 = load_nk_tile(Bt + 96, 128, lane);
        v8f c = {};
        c = wmma_step(a[0], b0, c);
        c = wmma_step(a[1], b1, c);
        c = wmma_step(a[2], b2, c);
        c = wmma_step(a[3], b3, c);

        int o = nt * 16 + n;                // output channel
        float bs = bo[o];
#pragma unroll
        for (int r = 0; r < 8; ++r) {
            int s  = mt * 16 + mbase + r;   // spatial = b*4096 + h*64 + w
            int bI = s >> 12;
            int hw = s & 4095;
            out[(((size_t)bI * Cch + o) << 12) + hw] = c[r] + bs;
        }
    }
}

// ---------------------------------------------------------------------------
// Launch
// ---------------------------------------------------------------------------
extern "C" void kernel_launch(void* const* d_in, const int* in_sizes, int n_in,
                              void* d_out, int out_size, void* d_ws, size_t ws_size,
                              hipStream_t stream) {
    (void)in_sizes; (void)n_in; (void)out_size; (void)ws_size;

    const float* x   = (const float*)d_in[0];
    const float* gw  = (const float*)d_in[1];
    const float* gb  = (const float*)d_in[2];
    const float* wq  = (const float*)d_in[3];
    const float* bq  = (const float*)d_in[4];
    const float* wk  = (const float*)d_in[5];
    const float* bk  = (const float*)d_in[6];
    const float* wv  = (const float*)d_in[7];
    const float* bv  = (const float*)d_in[8];
    const float* wo  = (const float*)d_in[9];
    const float* bo  = (const float*)d_in[10];
    const float* rpb = (const float*)d_in[11];
    float* out = (float*)d_out;

    // workspace carve-up (hipMalloc base is 256B aligned)
    char* w = (char*)d_ws;
    _Float16* wqh = (_Float16*)(w);                         // 128*128 f16 = 32 KB
    _Float16* wkh = wqh + 16384;
    _Float16* wvh = wkh + 16384;
    _Float16* woh = wvh + 16384;
    _Float16* xn  = (_Float16*)(w + 4 * 32768);             // 8192*128 f16 = 2 MB
    float* qb = (float*)(w + 4 * 32768 + 2097152);          // 8192*128 f32 = 4 MB
    float* kb = qb + (size_t)NSPAT * 128;
    float* vb = kb + (size_t)NSPAT * 128;
    _Float16* xo = (_Float16*)(vb + (size_t)NSPAT * 128);   // 2 MB

    convert_weights_kernel<<<64, 256, 0, stream>>>(wq, wk, wv, wo, wqh, wkh, wvh, woh);
    groupnorm_kernel<<<Bsz * 32, 256, 0, stream>>>(x, gw, gb, xn);
    qkv_proj_kernel<<<192, 256, 0, stream>>>(xn, wqh, wkh, wvh, bq, bk, bv, qb, kb, vb);
    natt_kernel<<<128, 256, 0, stream>>>(qb, kb, vb, rpb, xo);
    out_proj_kernel<<<64, 256, 0, stream>>>(xo, woh, bo, out);
}
